// RIMModule_50543175139713
// MI455X (gfx1250) — matrix operand order repaired and checked by reading, not compiled, not run on hardware
//
#include <hip/hip_runtime.h>
#include <hip/hip_bf16.h>

// ---------------------------------------------------------------------------
// RIM module forward for MI455X (gfx1250).
// All four GEMM stages run on v_wmma_f32_16x16x32_bf16 (f32 accumulate).
// bf16 operands precomputed once; wide tiles reuse each A fragment across
// up to 8 WMMAs per 32-wide k-chunk; LDS-repacked coalesced stores;
// WGP-scope global prefetch of streamed operands (locality=3).
// ---------------------------------------------------------------------------

typedef __bf16 bf16_t;
typedef __bf16 bf16x8  __attribute__((ext_vector_type(8)));
typedef __bf16 bf16x16 __attribute__((ext_vector_type(16)));
typedef float  f32x4   __attribute__((ext_vector_type(4)));
typedef float  f32x8   __attribute__((ext_vector_type(8)));

static constexpr int B_  = 32;
static constexpr int T_  = 1024;
static constexpr int D_  = 512;
static constexpr int K_  = 16;
static constexpr int H_  = 512;
static constexpr int TP  = 1056;      // T+1 padded to a multiple of 32
static constexpr int H3  = 3 * H_;    // 1536

#define DEV static __device__ __forceinline__

// float -> bf16, round-to-nearest-even (integer ops only)
DEV bf16_t f2bf(float f) {
  unsigned u = __builtin_bit_cast(unsigned, f);
  u += 0x7fffu + ((u >> 16) & 1u);
  unsigned short h = (unsigned short)(u >> 16);
  return __builtin_bit_cast(bf16_t, h);
}

DEV f32x8 zero8() {
  f32x8 z;
  for (int i = 0; i < 8; ++i) z[i] = 0.0f;
  return z;
}

DEV f32x8 wmma_bf16(bf16x16 a, bf16x16 b, f32x8 c) {
  // (neg_a, A, neg_b, B, c_mod, C, reuse_a, reuse_b)
  return __builtin_amdgcn_wmma_f32_16x16x32_bf16(false, a, false, b,
                                                 (short)0, c, false, false);
}

// A-fragment (16x32, lane=row): elements 0..15 = row[off..off+15];
// caller passes off = k0 + (lane>>4)*16
DEV bf16x16 afrag_bf16(const bf16_t* __restrict__ row, int off) {
  return *(const bf16x16*)(row + off);
}

// B-fragment (32x16, lane=column): contiguous 16 elements at k0 + (lane>>4)*16
DEV bf16x16 bfrag_bf16(const bf16_t* __restrict__ row, int off) {
  return *(const bf16x16*)(row + off);
}

// ---------------------------------------------------------------------------
// K0a: transpose + bf16-convert weights: Wk,Wv [D,H]->[H,D]; Wx,Wh [H,3H]->[3H,H]
// ---------------------------------------------------------------------------
__global__ void k_prep_weights(const float* __restrict__ Wk, const float* __restrict__ Wv,
                               const float* __restrict__ Wx, const float* __restrict__ Wh,
                               bf16_t* __restrict__ WkT, bf16_t* __restrict__ WvT,
                               bf16_t* __restrict__ WxT, bf16_t* __restrict__ WhT) {
  int idx = blockIdx.x * blockDim.x + threadIdx.x;
  if (idx < H_ * D_) {
    int h = idx / D_, d = idx % D_;
    WkT[idx] = f2bf(Wk[d * H_ + h]);
    WvT[idx] = f2bf(Wv[d * H_ + h]);
  }
  if (idx < H3 * H_) {
    int n = idx / H_, h = idx % H_;
    WxT[idx] = f2bf(Wx[h * H3 + n]);
    WhT[idx] = f2bf(Wh[h * H3 + n]);
  }
}

// ---------------------------------------------------------------------------
// K0b: bf16 activations, converted exactly once:
//  xcb [B][TP][D]  = bf16(x) with zero rows for t >= T (null/pad tokens)
//  qb  [B][K][H]   = bf16(hidden * q_map)
//  hidb[B][K][H]   = bf16(hidden)
// ---------------------------------------------------------------------------
__global__ void k_prep_acts(const float* __restrict__ x, const float* __restrict__ hid,
                            const float* __restrict__ qmap,
                            bf16_t* __restrict__ xcb, bf16_t* __restrict__ qb,
                            bf16_t* __restrict__ hidb) {
  size_t idx = (size_t)blockIdx.x * blockDim.x + threadIdx.x; // B*TP*D total
  int d = (int)(idx % D_);
  int r = (int)(idx / D_);
  int t = r % TP;
  int b = r / TP;
  float v = (t < T_) ? x[((size_t)b * T_ + t) * D_ + d] : 0.0f;
  xcb[idx] = f2bf(v);
  if (idx < (size_t)B_ * K_ * H_) {
    float hv = hid[idx];
    qb[idx]   = f2bf(hv * qmap[idx % (K_ * H_)]);
    hidb[idx] = f2bf(hv);
  }
}

// ---------------------------------------------------------------------------
// K1: keys/values projection. One wave per 16(t) x 64(h) tile computing BOTH
// keys and values (A fragment reused by 8 WMMAs per 32-wide k-chunk).
// keys stored [b][t][h] via LDS repack (coalesced 16B stores);
// values stored transposed [b][h][t] with packed 16B stores.
// ---------------------------------------------------------------------------
__global__ void k_kv_gemm(const bf16_t* __restrict__ xcb,
                          const bf16_t* __restrict__ WkT, const bf16_t* __restrict__ WvT,
                          const float* __restrict__ bk, const float* __restrict__ bv,
                          bf16_t* __restrict__ keys, bf16_t* __restrict__ valsT) {
  const int lane = threadIdx.x & 31;
  const int wv   = threadIdx.x >> 5;
  const int nTT = TP / 16, nTH = H_ / 64;       // 66 x 8
  int tile = blockIdx.x * 4 + wv;
  int b  = tile / (nTT * nTH);
  int r  = tile % (nTT * nTH);
  int t0 = (r / nTH) * 16;
  int h0 = (r % nTH) * 64;

  const int mn    = lane & 15;
  const int half  = lane >> 4;
  const int kaoff = half * 16;
  const int kboff = half * 16;

  const bf16_t* arow = xcb + ((size_t)b * TP + t0 + mn) * D_;
  const bf16_t* wk0  = WkT + (size_t)(h0 + mn) * D_;
  const bf16_t* wv0  = WvT + (size_t)(h0 + mn) * D_;

  f32x8 ck[4], cv[4];
#pragma unroll
  for (int j = 0; j < 4; ++j) { ck[j] = zero8(); cv[j] = zero8(); }

  for (int d0 = 0; d0 < D_; d0 += 32) {
    __builtin_prefetch((const void*)(arow + d0 + 64), 0, 3);  // WGP-scope
    bf16x16 a = afrag_bf16(arow, d0 + kaoff);
#pragma unroll
    for (int j = 0; j < 4; ++j) {
      const size_t rofs = (size_t)j * 16 * D_ + d0 + kboff;
      ck[j] = wmma_bf16(a, bfrag_bf16(wk0, (int)rofs), ck[j]);
      cv[j] = wmma_bf16(a, bfrag_bf16(wv0, (int)rofs), cv[j]);
    }
  }

  // ---- values: lane holds 8 contiguous t for column h0+j*16+mn -> packed
#pragma unroll
  for (int j = 0; j < 4; ++j) {
    float vb = bv[h0 + j * 16 + mn];
    bf16x8 pv;
#pragma unroll
    for (int rr = 0; rr < 8; ++rr) pv[rr] = f2bf(cv[j][rr] + vb);
    *(bf16x8*)(valsT + ((size_t)b * H_ + h0 + j * 16 + mn) * TP + t0 + 8 * half) = pv;
  }

  // ---- keys: repack 16(t) x 64(h) tile through LDS, then coalesced stores
  __shared__ bf16_t kt[4][16 * 64];
  bf16_t* slab = kt[wv];
#pragma unroll
  for (int j = 0; j < 4; ++j) {
    float kb = bk[h0 + j * 16 + mn];
    int col = j * 16 + mn;
#pragma unroll
    for (int rr = 0; rr < 8; ++rr) {
      slab[(8 * half + rr) * 64 + col] = f2bf(ck[j][rr] + kb);
    }
  }
  __syncthreads();
#pragma unroll
  for (int it = 0; it < 4; ++it) {
    int chunk = it * 32 + lane;          // 128 chunks of 8 bf16 (16B)
    int row = chunk >> 3;
    int cc  = chunk & 7;
    bf16x8 v = *(const bf16x8*)(slab + row * 64 + cc * 8);
    *(bf16x8*)(keys + ((size_t)b * TP + t0 + row) * H_ + h0 + cc * 8) = v;
  }
}

// ---------------------------------------------------------------------------
// K2: scores[b][k][t] = queries[k,:] . keys[t,:] * 1/sqrt(H)
// One wave per 16(k) x 32(t) tile, K-loop over H.
// ---------------------------------------------------------------------------
__global__ void k_scores(const bf16_t* __restrict__ qb, const bf16_t* __restrict__ keys,
                         float* __restrict__ scores) {
  const int lane = threadIdx.x & 31;
  const int wv   = threadIdx.x >> 5;
  const int nTT  = TP / 32;                    // 33
  int tile = blockIdx.x * 4 + wv;
  int b  = tile / nTT;
  int t0 = (tile % nTT) * 32;

  const int mn    = lane & 15;
  const int half  = lane >> 4;
  const int kaoff = half * 16;
  const int kboff = half * 16;
  const bf16_t* qrow = qb + ((size_t)b * K_ + mn) * H_;
  const bf16_t* k0   = keys + ((size_t)b * TP + t0 + mn) * H_;
  const bf16_t* k1   = k0 + (size_t)16 * H_;

  f32x8 c0 = zero8(), c1 = zero8();
  for (int h0 = 0; h0 < H_; h0 += 32) {
    __builtin_prefetch((const void*)(k0 + h0 + 64), 0, 3);
    bf16x16 a = afrag_bf16(qrow, h0 + kaoff);
    c0 = wmma_bf16(a, bfrag_bf16(k0, h0 + kboff), c0);
    c1 = wmma_bf16(a, bfrag_bf16(k1, h0 + kboff), c1);
  }
  const float scale = 0.04419417382415922f;    // 1/sqrt(512)
#pragma unroll
  for (int rr = 0; rr < 8; ++rr) {
    size_t base = ((size_t)b * K_ + rr + 8 * half) * TP + t0 + mn;
    scores[base]      = c0[rr] * scale;
    scores[base + 16] = c1[rr] * scale;
  }
}

// ---------------------------------------------------------------------------
// K3: softmax over t in [0, T] (1025 entries) per (b,k); attn written as bf16
// zero-padded to TP; captures null-token attention (t == T).
// ---------------------------------------------------------------------------
__global__ void k_softmax(const float* __restrict__ scores, bf16_t* __restrict__ attn,
                          float* __restrict__ nullAttn) {
  const int bk  = blockIdx.x;
  const int tid = threadIdx.x;                 // 256 threads
  const int NT  = T_ + 1;                      // 1025
  const float* s = scores + (size_t)bk * TP;
  bf16_t* arow   = attn + (size_t)bk * TP;
  __shared__ float red[256];

  float mx = -3.4e38f;
  for (int t = tid; t < NT; t += 256) mx = fmaxf(mx, s[t]);
  red[tid] = mx;
  __syncthreads();
  for (int st = 128; st > 0; st >>= 1) {
    if (tid < st) red[tid] = fmaxf(red[tid], red[tid + st]);
    __syncthreads();
  }
  mx = red[0];
  __syncthreads();

  float sum = 0.0f;
  for (int t = tid; t < NT; t += 256) sum += expf(s[t] - mx);
  red[tid] = sum;
  __syncthreads();
  for (int st = 128; st > 0; st >>= 1) {
    if (tid < st) red[tid] += red[tid + st];
    __syncthreads();
  }
  float inv = 1.0f / red[0];

  for (int t = tid; t < TP; t += 256) {
    float v = (t < NT) ? expf(s[t] - mx) * inv : 0.0f;
    arow[t] = f2bf(v);
  }
  if (tid == 0) nullAttn[bk] = expf(s[T_] - mx) * inv;
}

// ---------------------------------------------------------------------------
// K4: per batch, mark the active_k kernels with the SMALLEST null attention
// (first-index tie-breaking, matching lax.top_k on negated values).
// ---------------------------------------------------------------------------
__global__ void k_mask(const float* __restrict__ nullAttn, const int* __restrict__ activeK,
                       float* __restrict__ mask) {
  int b = blockIdx.x;
  if (threadIdx.x != 0) return;
  int ak = activeK[0];
  if (ak < 0) ak = 0;
  if (ak > K_) ak = K_;
  float v[K_];
  bool  sel[K_];
  for (int k = 0; k < K_; ++k) { v[k] = nullAttn[b * K_ + k]; sel[k] = false; }
  for (int it = 0; it < ak; ++it) {
    int best = -1;
    float bvv = 3.4e38f;
    for (int k = 0; k < K_; ++k)
      if (!sel[k] && v[k] < bvv) { bvv = v[k]; best = k; }
    if (best >= 0) sel[best] = true;
  }
  for (int k = 0; k < K_; ++k) mask[b * K_ + k] = sel[k] ? 1.0f : 0.0f;
}

// ---------------------------------------------------------------------------
// K5: attended[b][k][h] = sum_t attn[b][k][t] * values[b][t][h]
// One wave per 16(k) x 32(h) tile, K-loop over TP (attn zero-padded).
// ---------------------------------------------------------------------------
__global__ void k_attended(const bf16_t* __restrict__ attn, const bf16_t* __restrict__ valsT,
                           bf16_t* __restrict__ att) {
  const int lane = threadIdx.x & 31;
  const int wv   = threadIdx.x >> 5;
  const int nTH  = H_ / 32;                    // 16
  int tile = blockIdx.x * 4 + wv;
  int b  = tile / nTH;
  int h0 = (tile % nTH) * 32;

  const int mn    = lane & 15;
  const int half  = lane >> 4;
  const int kaoff = half * 16;
  const int kboff = half * 16;
  const bf16_t* arow = attn + ((size_t)b * K_ + mn) * TP;
  const bf16_t* v0   = valsT + ((size_t)b * H_ + h0 + mn) * TP;
  const bf16_t* v1   = v0 + (size_t)16 * TP;

  f32x8 c0 = zero8(), c1 = zero8();
  for (int t0 = 0; t0 < TP; t0 += 32) {
    __builtin_prefetch((const void*)(v0 + t0 + 64), 0, 3);
    bf16x16 a = afrag_bf16(arow, t0 + kaoff);
    c0 = wmma_bf16(a, bfrag_bf16(v0, t0 + kboff), c0);
    c1 = wmma_bf16(a, bfrag_bf16(v1, t0 + kboff), c1);
  }
#pragma unroll
  for (int rr = 0; rr < 8; ++rr) {
    size_t base = ((size_t)b * K_ + rr + 8 * half) * H_ + h0 + mn;
    att[base]      = f2bf(c0[rr]);
    att[base + 16] = f2bf(c1[rr]);
  }
}

// ---------------------------------------------------------------------------
// K6: GRU input gates. which==0: gx = attended @ Wx + bx
//                     which==1: gh = hidden   @ Wh + bh
// One wave per 16(k) x 64(n) tile over N = 3H, K-loop over H; A reused by
// 4 WMMAs per chunk; f32 output repacked through LDS for coalesced stores.
// ---------------------------------------------------------------------------
__global__ void k_gates(const bf16_t* __restrict__ att, const bf16_t* __restrict__ hidb,
                        const bf16_t* __restrict__ WxT, const bf16_t* __restrict__ WhT,
                        const float* __restrict__ bx, const float* __restrict__ bh,
                        float* __restrict__ gx, float* __restrict__ gh) {
  const int lane = threadIdx.x & 31;
  const int wv   = threadIdx.x >> 5;
  const int nTN  = H3 / 64;                    // 24
  int tile  = blockIdx.x * 4 + wv;
  int which = tile / (B_ * nTN);
  int r2    = tile % (B_ * nTN);
  int b  = r2 / nTN;
  int n0 = (r2 % nTN) * 64;

  const int mn    = lane & 15;
  const int half  = lane >> 4;
  const int kaoff = half * 16;
  const int kboff = half * 16;

  const bf16_t* wrow = (which ? WhT : WxT) + (size_t)(n0 + mn) * H_;
  const bf16_t* arow = (which ? hidb : att) + ((size_t)b * K_ + mn) * H_;
  const float*  bias = which ? bh : bx;
  float* out = which ? gh : gx;

  f32x8 c[4];
#pragma unroll
  for (int j = 0; j < 4; ++j) c[j] = zero8();

  for (int h0 = 0; h0 < H_; h0 += 32) {
    __builtin_prefetch((const void*)(wrow + h0 + 64), 0, 3);
    bf16x16 a = afrag_bf16(arow, h0 + kaoff);
#pragma unroll
    for (int j = 0; j < 4; ++j) {
      c[j] = wmma_bf16(a, bfrag_bf16(wrow, j * 16 * H_ + h0 + kboff), c[j]);
    }
  }

  // repack 16(k) x 64(n) f32 tile via LDS -> coalesced 16B stores
  __shared__ float gt[4][16 * 64];
  float* slab = gt[wv];
#pragma unroll
  for (int j = 0; j < 4; ++j) {
    float bb = bias[n0 + j * 16 + mn];
    int col = j * 16 + mn;
#pragma unroll
    for (int rr = 0; rr < 8; ++rr) {
      slab[(8 * half + rr) * 64 + col] = c[j][rr] + bb;
    }
  }
  __syncthreads();
#pragma unroll
  for (int it = 0; it < 8; ++it) {
    int chunk = it * 32 + lane;          // 256 chunks of 4 f32 (16B)
    int row = chunk >> 4;
    int cc  = chunk & 15;
    f32x4 v = *(const f32x4*)(slab + row * 64 + cc * 4);
    *(f32x4*)(out + ((size_t)b * K_ + row) * H3 + n0 + cc * 4) = v;
  }
}

// ---------------------------------------------------------------------------
// K7: GRU gates + masked blend -> d_out [B][K][H] f32
// ---------------------------------------------------------------------------
__global__ void k_gru(const float* __restrict__ gx, const float* __restrict__ gh,
                      const float* __restrict__ hid, const float* __restrict__ mask,
                      float* __restrict__ out) {
  size_t idx = (size_t)blockIdx.x * blockDim.x + threadIdx.x; // B*K*H total
  int h  = (int)(idx % H_);
  int bk = (int)(idx / H_);
  size_t g = (size_t)bk * H3;
  float r  = 1.0f / (1.0f + expf(-(gx[g + h] + gh[g + h])));
  float z  = 1.0f / (1.0f + expf(-(gx[g + H_ + h] + gh[g + H_ + h])));
  float n  = tanhf(gx[g + 2 * H_ + h] + r * gh[g + 2 * H_ + h]);
  float hp = hid[(size_t)bk * H_ + h];
  float nh = (1.0f - z) * n + z * hp;
  float m  = mask[bk];
  out[(size_t)bk * H_ + h] = m * nh + (1.0f - m) * hp;
}

// ---------------------------------------------------------------------------
extern "C" void kernel_launch(void* const* d_in, const int* in_sizes, int n_in,
                              void* d_out, int out_size, void* d_ws, size_t ws_size,
                              hipStream_t stream) {
  (void)in_sizes; (void)n_in; (void)out_size; (void)ws_size;
  const float* x    = (const float*)d_in[0];
  const float* hid  = (const float*)d_in[1];
  const float* qmap = (const float*)d_in[2];
  const float* Wk   = (const float*)d_in[3];
  const float* bk   = (const float*)d_in[4];
  const float* Wv   = (const float*)d_in[5];
  const float* bv   = (const float*)d_in[6];
  const float* Wx   = (const float*)d_in[7];
  const float* Wh   = (const float*)d_in[8];
  const float* bx   = (const float*)d_in[9];
  const float* bh   = (const float*)d_in[10];
  const int* activeK = (const int*)d_in[11];
  float* out = (float*)d_out;

  size_t off = 0;
  auto carve = [&](size_t bytes) -> char* {
    char* p = (char*)d_ws + off;
    off += (bytes + 255) & ~(size_t)255;
    return p;
  };
  bf16_t* WkT      = (bf16_t*)carve((size_t)H_ * D_ * sizeof(bf16_t));
  bf16_t* WvT      = (bf16_t*)carve((size_t)H_ * D_ * sizeof(bf16_t));
  bf16_t* WxT      = (bf16_t*)carve((size_t)H3 * H_ * sizeof(bf16_t));
  bf16_t* WhT      = (bf16_t*)carve((size_t)H3 * H_ * sizeof(bf16_t));
  bf16_t* xcb      = (bf16_t*)carve((size_t)B_ * TP * D_ * sizeof(bf16_t));
  bf16_t* qb       = (bf16_t*)carve((size_t)B_ * K_ * H_ * sizeof(bf16_t));
  bf16_t* hidb     = (bf16_t*)carve((size_t)B_ * K_ * H_ * sizeof(bf16_t));
  bf16_t* keys     = (bf16_t*)carve((size_t)B_ * TP * H_ * sizeof(bf16_t));
  bf16_t* valsT    = (bf16_t*)carve((size_t)B_ * H_ * TP * sizeof(bf16_t));
  float*  scores   = (float*) carve((size_t)B_ * K_ * TP * sizeof(float));
  bf16_t* attn     = (bf16_t*)carve((size_t)B_ * K_ * TP * sizeof(bf16_t));
  bf16_t* att      = (bf16_t*)carve((size_t)B_ * K_ * H_ * sizeof(bf16_t));
  float*  gx       = (float*) carve((size_t)B_ * K_ * H3 * sizeof(float));
  float*  gh       = (float*) carve((size_t)B_ * K_ * H3 * sizeof(float));
  float*  nullAttn = (float*) carve((size_t)B_ * K_ * sizeof(float));
  float*  mask     = (float*) carve((size_t)B_ * K_ * sizeof(float));

  k_prep_weights<<<(H3 * H_ + 255) / 256, 256, 0, stream>>>(Wk, Wv, Wx, Wh,
                                                            WkT, WvT, WxT, WhT);
  k_prep_acts<<<((size_t)B_ * TP * D_) / 256, 256, 0, stream>>>(x, hid, qmap,
                                                                xcb, qb, hidb);
  k_kv_gemm<<<(B_ * (TP / 16) * (H_ / 64)) / 4, 128, 0, stream>>>(
      xcb, WkT, WvT, bk, bv, keys, valsT);
  k_scores<<<(B_ * (TP / 32)) / 4, 128, 0, stream>>>(qb, keys, scores);
  k_softmax<<<B_ * K_, 256, 0, stream>>>(scores, attn, nullAttn);
  k_mask<<<B_, 32, 0, stream>>>(nullAttn, activeK, mask);
  k_attended<<<(B_ * (H_ / 32)) / 4, 128, 0, stream>>>(attn, valsT, att);
  k_gates<<<(2 * B_ * (H3 / 64)) / 4, 128, 0, stream>>>(att, hidb, WxT, WhT,
                                                        bx, bh, gx, gh);
  k_gru<<<(B_ * K_ * H_) / 256, 256, 0, stream>>>(gx, gh, hid, mask, out);
}
